// AttentionBlock_33887291965459
// MI455X (gfx1250) — matrix operand order repaired and checked by reading, not compiled, not run on hardware
//
#include <hip/hip_runtime.h>

typedef __attribute__((ext_vector_type(16))) _Float16 v16h;
typedef __attribute__((ext_vector_type(8)))  float    v8f;
typedef int v4i __attribute__((vector_size(4 * sizeof(int))));
typedef __attribute__((address_space(1))) v4i* gv4i_ptr;   // global
typedef __attribute__((address_space(3))) v4i* sv4i_ptr;   // LDS

#define WMMA_F16(a, b, c) \
  __builtin_amdgcn_wmma_f32_16x16x32_f16(false, (a), false, (b), (short)0, (c), false, false)

// ---- async global->LDS staging (CDNA5 GLOBAL_LOAD_ASYNC_TO_LDS_B128) -------
#if __has_builtin(__builtin_amdgcn_global_load_async_to_lds_b128)
#define ASYNC_STAGING 1
#define COPY16(dst, src)                                   \
  __builtin_amdgcn_global_load_async_to_lds_b128(          \
      (gv4i_ptr)(src), (sv4i_ptr)(dst), 0, 0)
#else
#define ASYNC_STAGING 0
#define COPY16(dst, src) (*(uint4*)(dst) = *(const uint4*)(src))
#endif

__device__ __forceinline__ void wait_async_zero() {
#if ASYNC_STAGING
#if __has_builtin(__builtin_amdgcn_s_wait_asynccnt)
  __builtin_amdgcn_s_wait_asynccnt(0);
#else
  asm volatile("s_wait_asynccnt 0x0" ::: "memory");
#endif
#endif
}

// 16-bit A-matrix 16x32 layout (ISA 7.12.2): lane L (L<16) row M=L holds
// K = {0..7,16..23}; lane L+16 holds K = {8..15,24..31}.
__device__ __forceinline__ int a_kidx(int i, int hi) {
  return (i < 8) ? (i + 8 * hi) : (16 + (i - 8) + 8 * hi);
}
// A fragment: lane holds row M=lane&15; halves K = i + 16*(lane>>4) (matches B).
__device__ __forceinline__ v16h load_a_frag(const _Float16* row, int k0, int hi) {
  v16h a;
#pragma unroll
  for (int i = 0; i < 16; ++i) a[i] = row[k0 + i + 16 * hi];
  return a;
}
// B fragment (32x16): lane holds column N=lane&15; halves K = i + 16*(lane>>4).
__device__ __forceinline__ v16h load_b_frag(const _Float16* row, int k0, int hi) {
  v16h b;
#pragma unroll
  for (int i = 0; i < 16; ++i) b[i] = row[k0 + i + 16 * hi];
  return b;
}

// ---------------------------------------------------------------------------
// Kernel 1: QKV projection.  qkv[b,t,f] = sum_e x[b,e,t] * qkv_w[f,e] + b[f]
// 32 tokens x 64 f per block (8 waves = 2 token-subtiles x 4 f-subtiles).
// Q pre-scaled by 1/sqrt(E)=1/16.  V is written TRANSPOSED: Vt[b][e][t],
// which makes the V store a contiguous 16B/lane store and lets the flash
// kernel stage V by pure byte copy (async-to-LDS friendly).
// ---------------------------------------------------------------------------
__global__ __launch_bounds__(256)
void qkv_proj_kernel(const float* __restrict__ x, const float* __restrict__ qkv_w,
                     const float* __restrict__ qkv_b,
                     _Float16* __restrict__ Q, _Float16* __restrict__ K,
                     _Float16* __restrict__ Vt) {
  const int N = 4096, E = 256;
  const int fg  = blockIdx.x;           // 12 groups of 64 f
  const int t0  = blockIdx.y * 32;      // 128 token tiles
  const int b   = blockIdx.z;
  const int tid = threadIdx.x;
  const int lane = tid & 31, w = tid >> 5;
  const int hi = lane >> 4, l16 = lane & 15;
  const int ti = w >> 2, fi = w & 3;

  __shared__ __attribute__((aligned(16))) _Float16 Xs[32][264];
  __shared__ __attribute__((aligned(16))) _Float16 Ws[64][264];

  { // X tile: 32 tokens x 256 e, gather along t (coalesced over lanes)
    const int e = tid >> 5, t = tid & 31;
    const float* xb = x + (size_t)b * E * N + t0 + t;
#pragma unroll
    for (int e0 = 0; e0 < 256; e0 += 8)
      Xs[t][e0 + e] = (_Float16)xb[(size_t)(e0 + e) * N];
  }
  { // W slice: 64 rows x 256, coalesced float4 + f16 convert
    const float4* wsrc = (const float4*)(qkv_w + (size_t)fg * 64 * 256);
    for (int i = tid; i < 64 * 64; i += 256) {
      const int r = i >> 6, c4 = (i & 63) << 2;
      float4 v = wsrc[i];
      Ws[r][c4 + 0] = (_Float16)v.x; Ws[r][c4 + 1] = (_Float16)v.y;
      Ws[r][c4 + 2] = (_Float16)v.z; Ws[r][c4 + 3] = (_Float16)v.w;
    }
  }
  __syncthreads();

  v8f c = {};
#pragma unroll
  for (int e0 = 0; e0 < 256; e0 += 32) {
    v16h a  = load_a_frag(&Xs[ti * 16 + l16][0], e0, hi);
    v16h bm = load_b_frag(&Ws[fi * 16 + l16][0], e0, hi);
    c = WMMA_F16(a, bm, c);
  }

  const int f = fg * 64 + fi * 16 + l16;
  const float bias = qkv_b[f];
  const int tb = t0 + ti * 16 + 8 * hi;     // C rows i=0..7 are consecutive t
  if (f < 256) {
#pragma unroll
    for (int i = 0; i < 8; ++i)
      Q[(size_t)(b * N + tb + i) * 256 + f] = (_Float16)((c[i] + bias) * 0.0625f);
  } else if (f < 512) {
    const int fo = f - 256;
#pragma unroll
    for (int i = 0; i < 8; ++i)
      K[(size_t)(b * N + tb + i) * 256 + fo] = (_Float16)(c[i] + bias);
  } else {
    const int fo = f - 512;
    __attribute__((aligned(16))) _Float16 tmp[8];
#pragma unroll
    for (int i = 0; i < 8; ++i) tmp[i] = (_Float16)(c[i] + bias);
    *(uint4*)&Vt[(size_t)(b * 256 + fo) * N + tb] = *(const uint4*)tmp;
  }
}

// ---------------------------------------------------------------------------
// Kernel 2: flash attention.  32 queries per block, stream 32-key chunks.
// Q/K/V staged via async global->LDS copies; V arrives pre-transposed so every
// WMMA B-fragment is a contiguous 32B LDS read.
// ---------------------------------------------------------------------------
__global__ __launch_bounds__(256)
void flash_attn_kernel(const _Float16* __restrict__ Q, const _Float16* __restrict__ K,
                       const _Float16* __restrict__ Vw, _Float16* __restrict__ O) {
  const int N = 4096;
  const int t0 = blockIdx.x * 32;
  const int b  = blockIdx.y;
  const int tid = threadIdx.x;
  const int lane = tid & 31, w = tid >> 5;
  const int hi = lane >> 4, l16 = lane & 15;

  __shared__ __attribute__((aligned(16))) _Float16 Qs[32][264];
  __shared__ __attribute__((aligned(16))) _Float16 Ks[32][264];
  __shared__ __attribute__((aligned(16))) _Float16 Vt[256][40];  // [e][k]
  __shared__ __attribute__((aligned(16))) float    Ss[32][36];
  __shared__ __attribute__((aligned(16))) _Float16 Ps[32][40];
  __shared__ float rowM[32], rowL[32], rowA[32];

  { // Q tile once: 32 x 256 halfs = 1024 x 16B
    const _Float16* qsrc = Q + (size_t)(b * N + t0) * 256;
    for (int i = tid; i < 1024; i += 256) {
      const int r = i >> 5, c = (i & 31) << 3;
      COPY16(&Qs[r][c], qsrc + i * 8);
    }
  }
  if (tid < 32) { rowM[tid] = -3.0e38f; rowL[tid] = 0.0f; }

  const int qi  = w & 1;   // S row tile (waves 0..3)
  const int ki  = w >> 1;  // S col tile (waves 0..3)
  const int qi2 = w >> 2;  // O row tile
  const int ci  = w & 3;   // O col group (64 cols)

  const _Float16* vbase = Vw + (size_t)b * 256 * N;   // Vw is [b][e][t]
  v8f oacc[4] = {{}, {}, {}, {}};

  for (int k0 = 0; k0 < N; k0 += 32) {
    { // stage K chunk [32][256] and pre-transposed V chunk [256][32]
      const _Float16* ksrc = K + (size_t)(b * N + k0) * 256;
      for (int i = tid; i < 1024; i += 256) {
        const int r = i >> 5, c = (i & 31) << 3;
        COPY16(&Ks[r][c], ksrc + i * 8);
        const int e = i >> 2, tc = (i & 3) << 3;
        COPY16(&Vt[e][tc], vbase + (size_t)e * N + k0 + tc);
      }
    }
    if (k0 + 32 < N) { // speculative prefetch of next chunk -> global_prefetch_b8
      __builtin_prefetch(K + (size_t)(b * N + k0 + 32) * 256 + tid * 32, 0, 1);
      __builtin_prefetch(vbase + (size_t)tid * N + k0 + 32, 0, 1);
    }
    wait_async_zero();
    __syncthreads();

    if (w < 4) { // S = Q K^T  (Q already carries 1/sqrt(E))
      v8f s = {};
#pragma unroll
      for (int e0 = 0; e0 < 256; e0 += 32) {
        v16h a  = load_a_frag(&Qs[qi * 16 + l16][0], e0, hi);
        v16h bm = load_b_frag(&Ks[ki * 16 + l16][0], e0, hi);
        s = WMMA_F16(a, bm, s);
      }
#pragma unroll
      for (int i = 0; i < 8; ++i)
        Ss[qi * 16 + i + 8 * hi][ki * 16 + l16] = s[i];
    }
    __syncthreads();

    if (tid < 32) { // online softmax, one row per thread
      const float mo = rowM[tid];
      float mx = mo;
#pragma unroll
      for (int j = 0; j < 32; ++j) mx = fmaxf(mx, Ss[tid][j]);
      const float alpha = __expf(mo - mx);
      float sum = 0.0f;
#pragma unroll
      for (int j = 0; j < 32; ++j) {
        const float p = __expf(Ss[tid][j] - mx);
        sum += p;
        Ps[tid][j] = (_Float16)p;
      }
      rowL[tid] = rowL[tid] * alpha + sum;
      rowM[tid] = mx;
      rowA[tid] = alpha;
    }
    __syncthreads();

    // rescale running O, then O += P @ V (K-dim 32 -> one WMMA per tile,
    // A-fragment reused across the 4 column tiles)
#pragma unroll
    for (int cc = 0; cc < 4; ++cc)
#pragma unroll
      for (int i = 0; i < 8; ++i)
        oacc[cc][i] *= rowA[qi2 * 16 + i + 8 * hi];

    v16h pa = load_a_frag(&Ps[qi2 * 16 + l16][0], 0, hi);
#pragma unroll
    for (int cc = 0; cc < 4; ++cc) {
      v16h vb = load_b_frag(&Vt[ci * 64 + cc * 16 + l16][0], 0, hi);
      oacc[cc] = WMMA_F16(pa, vb, oacc[cc]);
    }
    __syncthreads();
  }

  // normalize by running sum, store O[b,t,e] as f16
#pragma unroll
  for (int cc = 0; cc < 4; ++cc)
#pragma unroll
    for (int i = 0; i < 8; ++i) {
      const int t = t0 + qi2 * 16 + i + 8 * hi;
      const int e = ci * 64 + cc * 16 + l16;
      O[(size_t)(b * N + t) * 256 + e] =
          (_Float16)(oacc[cc][i] / rowL[qi2 * 16 + i + 8 * hi]);
    }
}

// ---------------------------------------------------------------------------
// Kernel 3: out projection + bias + residual, writes [b,E,h,w] layout.
// ---------------------------------------------------------------------------
__global__ __launch_bounds__(256)
void out_proj_kernel(const _Float16* __restrict__ O, const float* __restrict__ out_w,
                     const float* __restrict__ out_b, const float* __restrict__ x,
                     float* __restrict__ y) {
  const int N = 4096, E = 256;
  const int fg  = blockIdx.x;       // 4 groups of 64 f
  const int t0  = blockIdx.y * 32;  // 128 token tiles
  const int b   = blockIdx.z;
  const int tid = threadIdx.x;
  const int lane = tid & 31, w = tid >> 5;
  const int hi = lane >> 4, l16 = lane & 15;
  const int ti = w >> 2, fi = w & 3;

  __shared__ __attribute__((aligned(16))) _Float16 Os[32][264];
  __shared__ __attribute__((aligned(16))) _Float16 Ws[64][264];

  { // O tile via async byte copy
    const _Float16* osrc = O + (size_t)(b * N + t0) * 256;
    for (int i = tid; i < 1024; i += 256) {
      const int r = i >> 5, c = (i & 31) << 3;
      COPY16(&Os[r][c], osrc + i * 8);
    }
  }
  { // W slice with f32->f16 convert (VGPR path)
    const float4* wsrc = (const float4*)(out_w + (size_t)fg * 64 * 256);
    for (int i = tid; i < 64 * 64; i += 256) {
      const int r = i >> 6, c4 = (i & 63) << 2;
      float4 v = wsrc[i];
      Ws[r][c4 + 0] = (_Float16)v.x; Ws[r][c4 + 1] = (_Float16)v.y;
      Ws[r][c4 + 2] = (_Float16)v.z; Ws[r][c4 + 3] = (_Float16)v.w;
    }
  }
  wait_async_zero();
  __syncthreads();

  v8f c = {};
#pragma unroll
  for (int e0 = 0; e0 < 256; e0 += 32) {
    v16h a  = load_a_frag(&Os[ti * 16 + l16][0], e0, hi);
    v16h bm = load_b_frag(&Ws[fi * 16 + l16][0], e0, hi);
    c = WMMA_F16(a, bm, c);
  }

  const int f = fg * 64 + fi * 16 + l16;
  const float bias = out_b[f];
  const float* xr = x + (size_t)(b * E + f) * N;
  float*       yr = y + (size_t)(b * E + f) * N;
#pragma unroll
  for (int i = 0; i < 8; ++i) {
    const int t = t0 + ti * 16 + i + 8 * hi;
    yr[t] = xr[t] + bias + c[i];
  }
}

// ---------------------------------------------------------------------------
extern "C" void kernel_launch(void* const* d_in, const int* in_sizes, int n_in,
                              void* d_out, int out_size, void* d_ws, size_t ws_size,
                              hipStream_t stream) {
  const float* x     = (const float*)d_in[0];
  const float* qkv_w = (const float*)d_in[1];
  const float* qkv_b = (const float*)d_in[2];
  const float* out_w = (const float*)d_in[3];
  const float* out_b = (const float*)d_in[4];
  float* y = (float*)d_out;

  // workspace: Q,K [b][t][e], Vt [b][e][t], O [b][t][e]; all f16, 8 MB each
  const size_t per = (size_t)4 * 4096 * 256;
  _Float16* Q  = (_Float16*)d_ws;
  _Float16* K  = Q + per;
  _Float16* Vt = K + per;
  _Float16* O  = Vt + per;

  qkv_proj_kernel<<<dim3(12, 128, 4), 256, 0, stream>>>(x, qkv_w, qkv_b, Q, K, Vt);
  flash_attn_kernel<<<dim3(128, 4),    256, 0, stream>>>(Q, K, Vt, O);
  out_proj_kernel<<<dim3(4, 128, 4),   256, 0, stream>>>(O, out_w, out_b, x, y);
}
